// ModelWithgraph_embedding_73375221285171
// MI455X (gfx1250) — compile-verified, matrix operand order checked by reading, not run on hardware
//
#include <hip/hip_runtime.h>
#include <hip/hip_bf16.h>

// ---------------------------------------------------------------------------
// GNN forward on gfx1250 via v_wmma_f32_16x16x32_bf16.
//
//   h1 = relu(bn1(relu(segsum_dst( [x_d|x_s|ea|1] @ W1pad ))))      K=144->160
//   h2 = relu(bn2(relu(segsum_dst( [h1_d|h1_s|ea|1] @ W2pad ))))    K=272->288
//   ge = segsum_batch(h2)                                           [G=1024,128]
//   out = relu([h2|ge[batch]] @ fc1 + b1) @ fc2 + b2                [N,64]
//
// All GEMMs are 16-row wave tiles through WMMA bf16; linear biases are folded
// into an extra K row of the padded weight (A supplies 1.0 there).  The MP
// kernel is templated on <KIN_X, NKC> so every gather offset is a compile-time
// constant (branch-free A build, vectorized loads).
// ---------------------------------------------------------------------------

typedef __attribute__((ext_vector_type(16))) __bf16 v16bf;
typedef __attribute__((ext_vector_type(8)))  float  v8f;

#define GSEG 1024   // num graphs (reference G)

// 16-bit WMMA fragment element mapping (ISA 7.12.2, wave32):
// lane L: row = L & 15 ; element e (0..15): k = (e&7) + (L>>4)*8 + (e>>3)*16
__device__ __forceinline__ int kmap(int lane, int e) {
  return (e & 7) + ((lane >> 4) << 3) + ((e >> 3) << 4);
}

// ---------------------------------------------------------------------------
__global__ void fill_zero_kernel(float* __restrict__ p, long n) {
  long i = (long)blockIdx.x * blockDim.x + threadIdx.x;
  if (i < n) p[i] = 0.0f;
}

// Pack W [kin x nout] (+ optional bias as row kin) into fragment-major bf16:
// out[((kc*ntiles + nt)*32 + lane)*16 + e] = W[kc*32 + kmap(lane,e)][nt*16 + (lane&15)]
__global__ void pack_w_kernel(const float* __restrict__ W,
                              const float* __restrict__ bias,
                              int kin, int nout, int ntiles, int nkc,
                              __bf16* __restrict__ out) {
  int idx = blockIdx.x * blockDim.x + threadIdx.x;
  int total = nkc * ntiles * 512;
  if (idx >= total) return;
  int e    = idx & 15;
  int lane = (idx >> 4) & 31;
  int nt   = (idx >> 9) % ntiles;
  int kc   = idx / (512 * ntiles);
  int k = kc * 32 + kmap(lane, e);
  int n = nt * 16 + (lane & 15);
  float v = 0.0f;
  if (k < kin)                    v = W[(long)k * nout + n];
  else if (bias && k == kin)      v = bias[n];   // folded linear bias row
  out[idx] = (__bf16)v;
}

// ---------------------------------------------------------------------------
// Message-passing layer: one wave = 16 edges (N self-loops appended after the
// E real edges, edge_attr = 1.0).  A row = [x_d | x_s | ea | 1 | 0pad].
// Branch-free gather: edge index clamped (tail rows compute garbage that is
// never scattered); self-loop handled by selects.  Scatter-add via global
// fp32 atomics (segment_sum semantics), guarded only in the tail tile.
template <int KIN_X, int NKC>
__global__ __launch_bounds__(256) void mp_layer_kernel(
    const float* __restrict__ xin,
    const int* __restrict__ src, const int* __restrict__ dst,
    const float* __restrict__ ea, int nE, int nN,
    const __bf16* __restrict__ wfrag,
    float* __restrict__ hout) {
  constexpr int KIN2 = KIN_X * 2;
  const int lane = threadIdx.x & 31;
  const int wave = threadIdx.x >> 5;
  const int tile = blockIdx.x * (blockDim.x >> 5) + wave;
  const int ET = nE + nN;
  if (tile * 16 >= ET) return;   // wave-uniform

  const int r    = lane & 15;
  const int half = lane >> 4;
  const int eid  = tile * 16 + r;
  const int eidc  = (eid < ET) ? eid : (ET - 1);       // clamp (rows discarded)
  const bool self = eidc >= nE;
  const int eload = (eidc < nE) ? eidc : (nE - 1);     // safe src/dst/ea index
  const int sNode = self ? (eidc - nE) : src[eload];
  const int dNode = self ? (eidc - nE) : dst[eload];

  const float* rowD = xin + (long)dNode * KIN_X;
  const float* rowS = xin + (long)sNode * KIN_X;
  const float* rowE = ea + (long)eload * 16;

  v8f acc[8] = {};
#pragma unroll
  for (int kc = 0; kc < NKC; ++kc) {
    v16bf a;
#pragma unroll
    for (int e = 0; e < 16; ++e) {
      const int k = kc * 32 + (e & 7) + (half << 3) + ((e >> 3) << 4);
      float v;
      if (k < KIN_X) {
        v = rowD[k];
      } else if (k < KIN2) {
        v = rowS[k - KIN_X];
      } else if (k < KIN2 + 16) {
        float ev = rowE[k - KIN2];
        v = self ? 1.0f : ev;
      } else if (k == KIN2 + 16) {
        v = 1.0f;                       // bias row selector
      } else {
        v = 0.0f;
      }
      a[e] = (__bf16)v;
    }
#pragma unroll
    for (int nt = 0; nt < 8; ++nt) {
      const v16bf b = *(const v16bf*)(wfrag + ((long)(kc * 8 + nt) * 32 + lane) * 16);
      acc[nt] = __builtin_amdgcn_wmma_f32_16x16x32_bf16(
          false, a, false, b, (short)0, acc[nt], false, false);
    }
  }

  // scatter-add: D element v of lane L is row M = v + 8*half, col (L&15)+nt*16
  if (tile * 16 + 16 <= ET) {
    // full tile: no guards
#pragma unroll
    for (int v = 0; v < 8; ++v) {
      const int eid2 = tile * 16 + v + half * 8;
      const int d2 = (eid2 < nE) ? dst[eid2] : (eid2 - nE);
      float* row = hout + (long)d2 * 128 + (lane & 15);
#pragma unroll
      for (int nt = 0; nt < 8; ++nt)
        atomicAdd(row + nt * 16, acc[nt][v]);
    }
  } else {
#pragma unroll
    for (int v = 0; v < 8; ++v) {
      const int eid2 = tile * 16 + v + half * 8;
      if (eid2 >= ET) continue;
      const int d2 = (eid2 < nE) ? dst[eid2] : (eid2 - nE);
      float* row = hout + (long)d2 * 128 + (lane & 15);
#pragma unroll
      for (int nt = 0; nt < 8; ++nt)
        atomicAdd(row + nt * 16, acc[nt][v]);
    }
  }
}

// relu -> eval-BN (mean=0,var=1) -> relu, in place on [n] elems (c = i % 128)
__global__ void relu_bn_relu_kernel(float* __restrict__ h,
                                    const float* __restrict__ g,
                                    const float* __restrict__ b, long n) {
  long i = (long)blockIdx.x * blockDim.x + threadIdx.x;
  if (i >= n) return;
  int c = (int)(i & 127);
  float scale = g[c] * rsqrtf(1.0f + 1e-5f);
  float t = fmaxf(h[i], 0.0f);
  t = t * scale + b[c];
  h[i] = fmaxf(t, 0.0f);
}

// global_add_pool: ge[batch[node]] += h[node]
__global__ void seg_pool_kernel(const float* __restrict__ h,
                                const int* __restrict__ batch,
                                float* __restrict__ ge, int nN) {
  long i = (long)blockIdx.x * blockDim.x + threadIdx.x;
  if (i >= (long)nN * 128) return;
  int node = (int)(i >> 7);
  int c    = (int)(i & 127);
  atomicAdd(&ge[(long)batch[node] * 128 + c], h[i]);
}

// ---------------------------------------------------------------------------
// Final node MLP: one wave = 16 nodes. A = [h2 | ge[batch]] (K=256, 8 chunks).
// fc1 (16 N-tiles) -> bias+relu -> bf16 staged in per-wave LDS in row-major A[m][k]
// addressing -> fc2 (4 N-tiles) -> + bias -> out [N,64].
__global__ __launch_bounds__(128) void final_mlp_kernel(
    const float* __restrict__ h2, const float* __restrict__ ge,
    const int* __restrict__ batch,
    const __bf16* __restrict__ f1frag, const float* __restrict__ fc1_b,
    const __bf16* __restrict__ f2frag, const float* __restrict__ fc2_b,
    float* __restrict__ out, int nN) {
  __shared__ __bf16 hid[4][16 * 256];   // 32 KB, private per wave
  const int lane = threadIdx.x & 31;
  const int wave = threadIdx.x >> 5;
  const int tile = blockIdx.x * 4 + wave;
  if (tile * 16 >= nN) return;          // wave-uniform, no block barrier used

  const int r    = lane & 15;
  const int half = lane >> 4;
  const int node = tile * 16 + r;
  const int nodc = (node < nN) ? node : (nN - 1);   // clamp; tail rows discarded
  const int g = batch[nodc];
  const float* rowH = h2 + (long)nodc * 128;
  const float* rowG = ge + (long)g * 128;

  v8f acc[16] = {};
#pragma unroll
  for (int kc = 0; kc < 8; ++kc) {
    v16bf a;
#pragma unroll
    for (int e = 0; e < 16; ++e) {
      const int k = kc * 32 + (e & 7) + (half << 3) + ((e >> 3) << 4);
      float v = (k < 128) ? rowH[k] : rowG[k - 128];
      a[e] = (__bf16)v;
    }
#pragma unroll
    for (int nt = 0; nt < 16; ++nt) {
      const v16bf b = *(const v16bf*)(f1frag + ((long)(kc * 16 + nt) * 32 + lane) * 16);
      acc[nt] = __builtin_amdgcn_wmma_f32_16x16x32_bf16(
          false, a, false, b, (short)0, acc[nt], false, false);
    }
  }

  // bias + relu, stage hidden (bf16) into LDS addressed as A[r][k]
#pragma unroll
  for (int v = 0; v < 8; ++v) {
    const int m = v + half * 8;
#pragma unroll
    for (int nt = 0; nt < 16; ++nt) {
      const int col = nt * 16 + (lane & 15);
      float t = fmaxf(acc[nt][v] + fc1_b[col], 0.0f);
      hid[wave][m * 256 + col] = (__bf16)t;
    }
  }
  // same-wave LDS RAW is in-order; compiler inserts waits for the loads

  v8f acc2[4] = {};
#pragma unroll
  for (int kc = 0; kc < 8; ++kc) {
    v16bf a;
#pragma unroll
    for (int e = 0; e < 16; ++e)
      a[e] = hid[wave][r * 256 + kc * 32 + (e & 7) + (half << 3) + ((e >> 3) << 4)];
#pragma unroll
    for (int nt = 0; nt < 4; ++nt) {
      const v16bf b = *(const v16bf*)(f2frag + ((long)(kc * 4 + nt) * 32 + lane) * 16);
      acc2[nt] = __builtin_amdgcn_wmma_f32_16x16x32_bf16(
          false, a, false, b, (short)0, acc2[nt], false, false);
    }
  }

  if (tile * 16 + 16 <= nN) {
#pragma unroll
    for (int v = 0; v < 8; ++v) {
      const int n2 = tile * 16 + v + half * 8;
      float* orow = out + (long)n2 * 64 + (lane & 15);
#pragma unroll
      for (int nt = 0; nt < 4; ++nt)
        orow[nt * 16] = acc2[nt][v] + fc2_b[nt * 16 + (lane & 15)];
    }
  } else {
#pragma unroll
    for (int v = 0; v < 8; ++v) {
      const int n2 = tile * 16 + v + half * 8;
      if (n2 >= nN) continue;
      float* orow = out + (long)n2 * 64 + (lane & 15);
#pragma unroll
      for (int nt = 0; nt < 4; ++nt)
        orow[nt * 16] = acc2[nt][v] + fc2_b[nt * 16 + (lane & 15)];
    }
  }
}

// ---------------------------------------------------------------------------
extern "C" void kernel_launch(void* const* d_in, const int* in_sizes, int n_in,
                              void* d_out, int out_size, void* d_ws, size_t ws_size,
                              hipStream_t stream) {
  (void)n_in; (void)out_size; (void)ws_size;
  const float* x      = (const float*)d_in[0];
  const int*   src    = (const int*)  d_in[1];
  const int*   dst    = (const int*)  d_in[2];
  const float* ea     = (const float*)d_in[3];
  const int*   batch  = (const int*)  d_in[4];
  const float* lin1_w = (const float*)d_in[5];
  const float* lin1_b = (const float*)d_in[6];
  const float* lin2_w = (const float*)d_in[7];
  const float* lin2_b = (const float*)d_in[8];
  const float* bn1_g  = (const float*)d_in[9];
  const float* bn1_b  = (const float*)d_in[10];
  const float* bn2_g  = (const float*)d_in[11];
  const float* bn2_b  = (const float*)d_in[12];
  const float* fc1_w  = (const float*)d_in[13];
  const float* fc1_b  = (const float*)d_in[14];
  const float* fc2_w  = (const float*)d_in[15];
  const float* fc2_b  = (const float*)d_in[16];

  const int N = in_sizes[0] / 64;
  const int E = in_sizes[1];
  float* out = (float*)d_out;

  // workspace carve-up
  char* ws = (char*)d_ws;
  auto carve = [&](size_t bytes) -> char* {
    char* p = ws; ws += (bytes + 255) & ~(size_t)255; return p;
  };
  float*  h1  = (float*) carve((size_t)N * 128 * 4);
  float*  h2  = (float*) carve((size_t)N * 128 * 4);
  float*  gem = (float*) carve((size_t)GSEG * 128 * 4);
  __bf16* w1f = (__bf16*)carve((size_t)5 * 8  * 512 * 2);  // K=160
  __bf16* w2f = (__bf16*)carve((size_t)9 * 8  * 512 * 2);  // K=288
  __bf16* f1f = (__bf16*)carve((size_t)8 * 16 * 512 * 2);  // K=256, N=256
  __bf16* f2f = (__bf16*)carve((size_t)8 * 4  * 512 * 2);  // K=256, N=64

  // pack weights (bias folded as K row kin for the message MLPs)
  {
    int t1 = 5 * 8 * 512, t2 = 9 * 8 * 512, t3 = 8 * 16 * 512, t4 = 8 * 4 * 512;
    pack_w_kernel<<<(t1 + 255) / 256, 256, 0, stream>>>(lin1_w, lin1_b, 144, 128, 8,  5, w1f);
    pack_w_kernel<<<(t2 + 255) / 256, 256, 0, stream>>>(lin2_w, lin2_b, 272, 128, 8,  9, w2f);
    pack_w_kernel<<<(t3 + 255) / 256, 256, 0, stream>>>(fc1_w,  nullptr,256, 256, 16, 8, f1f);
    pack_w_kernel<<<(t4 + 255) / 256, 256, 0, stream>>>(fc2_w,  nullptr,256, 64,  4,  8, f2f);
  }

  const long nh = (long)N * 128;
  const int ET = E + N;
  const int tilesE  = (ET + 15) / 16;
  const int blocksE = (tilesE + 7) / 8;     // 8 waves / block

  // layer 1
  fill_zero_kernel<<<(int)((nh + 255) / 256), 256, 0, stream>>>(h1, nh);
  mp_layer_kernel<64, 5><<<blocksE, 256, 0, stream>>>(x, src, dst, ea, E, N, w1f, h1);
  relu_bn_relu_kernel<<<(int)((nh + 255) / 256), 256, 0, stream>>>(h1, bn1_g, bn1_b, nh);

  // layer 2
  fill_zero_kernel<<<(int)((nh + 255) / 256), 256, 0, stream>>>(h2, nh);
  mp_layer_kernel<128, 9><<<blocksE, 256, 0, stream>>>(h1, src, dst, ea, E, N, w2f, h2);
  relu_bn_relu_kernel<<<(int)((nh + 255) / 256), 256, 0, stream>>>(h2, bn2_g, bn2_b, nh);

  // graph pooling
  const long ng = (long)GSEG * 128;
  fill_zero_kernel<<<(int)((ng + 255) / 256), 256, 0, stream>>>(gem, ng);
  seg_pool_kernel<<<(int)((nh + 255) / 256), 256, 0, stream>>>(h2, batch, gem, N);

  // final node MLP
  const int tilesN  = (N + 15) / 16;
  const int blocksN = (tilesN + 3) / 4;     // 4 waves / block
  final_mlp_kernel<<<blocksN, 128, 0, stream>>>(h2, gem, batch, f1f, fc1_b, f2f, fc2_b, out, N);
}
